// MultiHeadAttention_41403484733727
// MI455X (gfx1250) — matrix operand order repaired and checked by reading, not compiled, not run on hardware
//
#include <hip/hip_runtime.h>

typedef __bf16 bf16_t;
typedef __attribute__((ext_vector_type(16))) __bf16 v16bf;
typedef __attribute__((ext_vector_type(8)))  float  v8f;
typedef __attribute__((ext_vector_type(4)))  unsigned int v4u;

union Frag { v16bf v; v4u q[2]; };

#define WMMA_BF16(A, B, C) \
  __builtin_amdgcn_wmma_f32_16x16x32_bf16(false, (A), false, (B), (short)0, (C), false, false)

// ---------------------------------------------------------------------------
// Prep: fp32 -> bf16 conversion
// ---------------------------------------------------------------------------
__global__ void cvt_f32_bf16(const float* __restrict__ x, bf16_t* __restrict__ y, int n) {
  int i = blockIdx.x * blockDim.x + threadIdx.x;
  if (i < n) y[i] = (bf16_t)x[i];
}

// Prep: WT[k*1024 + e] = (bf16) W[e*1024 + k]   (E == 1024)
__global__ void wtranspose(const float* __restrict__ W, bf16_t* __restrict__ WT) {
  int i = blockIdx.x * blockDim.x + threadIdx.x;  // i = k*1024 + e
  int k = i >> 10, e = i & 1023;
  WT[i] = (bf16_t)W[(size_t)e * 1024 + k];
}

// ---------------------------------------------------------------------------
// Tiled bf16 GEMM: C[M,N] = A[M,K] * Bm[K,N]  (Bm = W^T pre-transposed)
//   mode 0: store bf16 row-major
//   mode 1: store bf16 as Kt[b,h,d,s]  (per-head transposed; S=2048,H=16,D=64)
//   mode 2: store fp32 row-major + bias[n]
// Workgroup: 256 threads (8 waves), 128x128 tile, K-step 32,
// double-buffered LDS with software-pipelined global loads.
// ---------------------------------------------------------------------------
__global__ __launch_bounds__(256) void gemm_bf16(
    const bf16_t* __restrict__ A, const bf16_t* __restrict__ Bm,
    void* __restrict__ Cout, const float* __restrict__ bias,
    int M, int N, int K, int mode)
{
  __shared__ bf16_t As[2][128][40];   // padded stride (80B, 16B-aligned rows)
  __shared__ bf16_t Bs[2][32][136];   // padded stride (272B, 16B-aligned rows)

  const int tid   = threadIdx.x;
  const int lane  = tid & 31;
  const int wave  = tid >> 5;
  const int waveM = wave >> 1;     // 0..3 -> 32-row slab
  const int waveN = wave & 1;      // 0..1 -> 64-col slab
  const int m0 = blockIdx.y * 128;
  const int n0 = blockIdx.x * 128;

  v8f acc[2][4] = {};

  const int ra = tid >> 1, csa = (tid & 1) * 16;   // A staging: 32B per thread
  const int rb = tid >> 3, csb = (tid & 7) * 16;   // B staging: 32B per thread
  const bf16_t* pa = A  + (size_t)(m0 + ra) * K + csa;   // advance by k
  const bf16_t* pb = Bm + (size_t)rb * N + n0 + csb;     // advance by k*N

  const int nsteps = K >> 5;

  // prologue: stage step 0 into buffer 0
  {
    *(v4u*)&As[0][ra][csa]     = *(const v4u*)pa;
    *(v4u*)&As[0][ra][csa + 8] = *(const v4u*)(pa + 8);
    *(v4u*)&Bs[0][rb][csb]     = *(const v4u*)pb;
    *(v4u*)&Bs[0][rb][csb + 8] = *(const v4u*)(pb + 8);
  }
  __syncthreads();

  const int ca = (lane >> 4) << 3;
  const int rA0 = waveM * 32 + (lane & 15);

  for (int s = 0; s < nsteps; ++s) {
    const int cur = s & 1;

    // issue next tile's global loads early (overlap with WMMA below)
    v4u na0, na1, nb0, nb1;
    if (s + 1 < nsteps) {
      const bf16_t* sa = pa + (size_t)(s + 1) * 32;
      const bf16_t* sb = pb + (size_t)(s + 1) * 32 * N;
      na0 = *(const v4u*)sa;       na1 = *(const v4u*)(sa + 8);
      nb0 = *(const v4u*)sb;       nb1 = *(const v4u*)(sb + 8);
    }
    if (s + 2 < nsteps) {
      __builtin_prefetch(pa + (size_t)(s + 2) * 32, 0, 0);
      __builtin_prefetch(pb + (size_t)(s + 2) * 32 * N, 0, 0);
    }

    // A fragments: 16x32 bf16, element j of lane L holds
    //   K = (j>=8 ? 16:0) + (L>=16 ? 8:0) + (j&7), row M = L&15
    Frag a[2];
    for (int mt = 0; mt < 2; ++mt) {
      int r = rA0 + mt * 16;
      a[mt].q[0] = *(const v4u*)&As[cur][r][ca];
      a[mt].q[1] = *(const v4u*)&As[cur][r][ca + 16];
    }
    // B fragments: 32x16 bf16, column N = L&15, element j holds same K as A
    for (int nt = 0; nt < 4; ++nt) {
      Frag b;
      int c = waveN * 64 + nt * 16;
      for (int j = 0; j < 16; ++j)
        b.v[j] = Bs[cur][(j & 7) + ((lane >> 4) << 3) + ((j >> 3) << 4)][c + (lane & 15)];
      acc[0][nt] = WMMA_BF16(a[0].v, b.v, acc[0][nt]);
      acc[1][nt] = WMMA_BF16(a[1].v, b.v, acc[1][nt]);
    }

    // drain next tile into the alternate buffer
    if (s + 1 < nsteps) {
      const int nxt = cur ^ 1;
      *(v4u*)&As[nxt][ra][csa]     = na0;
      *(v4u*)&As[nxt][ra][csa + 8] = na1;
      *(v4u*)&Bs[nxt][rb][csb]     = nb0;
      *(v4u*)&Bs[nxt][rb][csb + 8] = nb1;
    }
    __syncthreads();
  }

  // Epilogue. D layout: VGPR v, lane -> M = v + 8*(lane>>4), N = lane&15
  for (int mt = 0; mt < 2; ++mt)
    for (int nt = 0; nt < 4; ++nt)
      for (int v = 0; v < 8; ++v) {
        int gm = m0 + waveM * 32 + mt * 16 + v + ((lane >> 4) << 3);
        int gn = n0 + waveN * 64 + nt * 16 + (lane & 15);
        float val = acc[mt][nt][v];
        if (mode == 0) {
          ((bf16_t*)Cout)[(size_t)gm * N + gn] = (bf16_t)val;
        } else if (mode == 1) {
          int b_ = gm >> 11, s_ = gm & 2047;     // S = 2048
          int h_ = gn >> 6,  d_ = gn & 63;       // D = 64
          ((bf16_t*)Cout)[(((size_t)b_ * 16 + h_) * 64 + d_) * 2048 + s_] = (bf16_t)val;
        } else {
          ((float*)Cout)[(size_t)gm * N + gn] = val + bias[gn];
        }
      }
}

// ---------------------------------------------------------------------------
// Sliding-window attention (window = +/-128), one WG per (b,h, 32-query tile).
// 128 threads = 4 waves. Key strip: fixed 320 columns [q0-128, q0+192);
// 16/32-wide key tiles are always fully inside or fully outside [0,S).
// ---------------------------------------------------------------------------
__global__ __launch_bounds__(128) void attn_local(
    const bf16_t* __restrict__ Q,   // [B,S,H,D] bf16 (row-major [B*S, E])
    const bf16_t* __restrict__ Kt,  // [B,H,D,S] bf16
    const bf16_t* __restrict__ V,   // [B,S,H,D] bf16
    bf16_t* __restrict__ AO)        // [B,S,H,D] bf16
{
  __shared__ float  sc[32][320];    // scores (fp32)
  __shared__ bf16_t pP[32][320];    // normalized probabilities (bf16)
  __shared__ float  red[32][4];     // softmax partial max/sum

  const int tid  = threadIdx.x;
  const int lane = tid & 31;
  const int w    = tid >> 5;            // wave 0..3
  const int q0   = blockIdx.x * 32;     // query tile start
  const int bh   = blockIdx.y;
  const int b    = bh >> 4, h = bh & 15;
  const int kstart = q0 - 128;          // multiple of 32

  // ---- phase A: S = Q K^T (each wave owns 5 of the 20 key tiles) ----
  v8f sacc[2][5] = {};
  for (int kk = 0; kk < 64; kk += 32) {
    Frag a[2];
    const int ca = kk + ((lane >> 4) << 3);
    for (int mt = 0; mt < 2; ++mt) {
      int s = q0 + mt * 16 + (lane & 15);
      const bf16_t* qp = Q + ((size_t)(b * 2048 + s)) * 1024 + h * 64;
      a[mt].q[0] = *(const v4u*)(qp + ca);
      a[mt].q[1] = *(const v4u*)(qp + ca + 16);
    }
    const bf16_t* kp = Kt + ((size_t)(bh * 64 + kk)) * 2048;
    for (int nti = 0; nti < 5; ++nti) {
      int g  = w * 5 + nti;
      int t0 = kstart + g * 16;
      if ((unsigned)t0 >= 2048u) continue;                 // wave-uniform skip
      Frag bfr;
      for (int j = 0; j < 16; ++j)
        bfr.v[j] = kp[(size_t)((j & 7) + ((lane >> 4) << 3) + ((j >> 3) << 4)) * 2048 + t0 + (lane & 15)];
      sacc[0][nti] = WMMA_BF16(a[0].v, bfr.v, sacc[0][nti]);
      sacc[1][nti] = WMMA_BF16(a[1].v, bfr.v, sacc[1][nti]);
    }
  }
  // scale (1/sqrt(64)) + sliding-window mask -> LDS
  for (int mt = 0; mt < 2; ++mt)
    for (int nti = 0; nti < 5; ++nti)
      for (int v = 0; v < 8; ++v) {
        int r = mt * 16 + v + ((lane >> 4) << 3);
        int c = (w * 5 + nti) * 16 + (lane & 15);
        int i = q0 + r, j = kstart + c;
        bool ok = (j >= 0) && (j < 2048) && (i - j <= 128) && (j - i <= 128);
        sc[r][c] = ok ? sacc[mt][nti][v] * 0.125f : -1e9f;
      }
  __syncthreads();

  // ---- softmax: 4 threads per row, 80-column segments ----
  {
    const int r = tid >> 2, seg = tid & 3;
    const int c0 = seg * 80, c1 = c0 + 80;
    float mx = -3.0e38f;
    for (int c = c0; c < c1; ++c) mx = fmaxf(mx, sc[r][c]);
    red[r][seg] = mx;
    __syncthreads();
    mx = fmaxf(fmaxf(red[r][0], red[r][1]), fmaxf(red[r][2], red[r][3]));
    __syncthreads();
    float sum = 0.f;
    for (int c = c0; c < c1; ++c) { float e = expf(sc[r][c] - mx); sc[r][c] = e; sum += e; }
    red[r][seg] = sum;
    __syncthreads();
    sum = (red[r][0] + red[r][1]) + (red[r][2] + red[r][3]);
    float inv = 1.0f / sum;
    for (int c = c0; c < c1; ++c) pP[r][c] = (bf16_t)(sc[r][c] * inv);
  }
  __syncthreads();

  // ---- phase B: O = P V (each wave owns 16 of the 64 head dims) ----
  const int d0 = w * 16;
  v8f oacc[2] = {};
  for (int ks = 0; ks < 10; ++ks) {
    int t0 = kstart + ks * 32;
    if ((unsigned)t0 >= 2048u) continue;                   // wave-uniform skip
    Frag a[2];
    int cc = ks * 32 + ((lane >> 4) << 3);
    for (int mt = 0; mt < 2; ++mt) {
      int r = mt * 16 + (lane & 15);
      a[mt].q[0] = *(const v4u*)&pP[r][cc];
      a[mt].q[1] = *(const v4u*)&pP[r][cc + 16];
    }
    Frag bfr;
    const bf16_t* vp = V + ((size_t)(b * 2048 + t0)) * 1024 + h * 64 + d0 + (lane & 15);
    for (int j = 0; j < 16; ++j)
      bfr.v[j] = vp[(size_t)((j & 7) + ((lane >> 4) << 3) + ((j >> 3) << 4)) * 1024];
    oacc[0] = WMMA_BF16(a[0].v, bfr.v, oacc[0]);
    oacc[1] = WMMA_BF16(a[1].v, bfr.v, oacc[1]);
  }
  for (int mt = 0; mt < 2; ++mt)
    for (int v = 0; v < 8; ++v) {
      int s = q0 + mt * 16 + v + ((lane >> 4) << 3);
      int d = d0 + (lane & 15);
      AO[((size_t)(b * 2048 + s)) * 1024 + h * 64 + d] = (bf16_t)oacc[mt][v];
    }
}

// ---------------------------------------------------------------------------
extern "C" void kernel_launch(void* const* d_in, const int* in_sizes, int n_in,
                              void* d_out, int out_size, void* d_ws, size_t ws_size,
                              hipStream_t stream)
{
  const float* x  = (const float*)d_in[0];
  const float* Wq = (const float*)d_in[1];
  const float* Wk = (const float*)d_in[2];
  const float* Wv = (const float*)d_in[3];
  const float* Wo = (const float*)d_in[4];
  const float* bo = (const float*)d_in[5];

  const int E = 1024, S = 2048;
  const int M  = in_sizes[0] / E;   // B*S = 4096
  const int Bn = M / S;             // batch = 2

  char* p = (char*)d_ws;
  bf16_t* xbf = (bf16_t*)p;  p += (size_t)M * E * 2;
  bf16_t* wqt = (bf16_t*)p;  p += (size_t)E * E * 2;
  bf16_t* wkt = (bf16_t*)p;  p += (size_t)E * E * 2;
  bf16_t* wvt = (bf16_t*)p;  p += (size_t)E * E * 2;
  bf16_t* wot = (bf16_t*)p;  p += (size_t)E * E * 2;
  bf16_t* qb  = (bf16_t*)p;  p += (size_t)M * E * 2;
  bf16_t* kt  = (bf16_t*)p;  p += (size_t)M * E * 2;
  bf16_t* vb  = (bf16_t*)p;  p += (size_t)M * E * 2;
  bf16_t* ao  = (bf16_t*)p;  p += (size_t)M * E * 2;

  const int nx = M * E;
  cvt_f32_bf16<<<(nx + 255) / 256, 256, 0, stream>>>(x, xbf, nx);
  const int nw = E * E;
  wtranspose<<<(nw + 255) / 256, 256, 0, stream>>>(Wq, wqt);
  wtranspose<<<(nw + 255) / 256, 256, 0, stream>>>(Wk, wkt);
  wtranspose<<<(nw + 255) / 256, 256, 0, stream>>>(Wv, wvt);
  wtranspose<<<(nw + 255) / 256, 256, 0, stream>>>(Wo, wot);

  dim3 gg(E / 128, M / 128);        // 8 x 32 workgroups
  gemm_bf16<<<gg, 256, 0, stream>>>(xbf, wqt, qb, nullptr, M, E, E, 0);  // Q
  gemm_bf16<<<gg, 256, 0, stream>>>(xbf, wkt, kt, nullptr, M, E, E, 1);  // K^T per head
  gemm_bf16<<<gg, 256, 0, stream>>>(xbf, wvt, vb, nullptr, M, E, E, 0);  // V

  attn_local<<<dim3(S / 32, Bn * 16), 128, 0, stream>>>(qb, kt, vb, ao);

  gemm_bf16<<<gg, 256, 0, stream>>>(ao, wot, d_out, bo, M, E, E, 2);     // O + bias
}
